// CausalAttentiveStatisticsPooling_63316407877638
// MI455X (gfx1250) — compile-verified, matrix operand order checked by reading, not run on hardware
//
#include <hip/hip_runtime.h>

#define Bq   8
#define Cq   512
#define Tq   2048
#define Aq   128
#define K3C  1536
#define NKS  48          // K3C / 32 k-steps
#define EPSq 1e-12f

typedef __attribute__((ext_vector_type(16))) __bf16 v16bf;
typedef __attribute__((ext_vector_type(8)))  __bf16 v8bf;
typedef __attribute__((ext_vector_type(8)))  float  v8f;

__device__ __forceinline__ unsigned short f2bf(float f) {
    unsigned u = __float_as_uint(f);
    u += 0x7FFFu + ((u >> 16) & 1u);          // round-to-nearest-even
    return (unsigned short)(u >> 16);
}

// fragment coordinates for 16-bit operand layout (ISA 7.12.2):
// k%32 -> (half, vgpr, slot)
__device__ __forceinline__ void fragcoord(int kl, int& hh, int& slot) {
    int vv;
    if (kl < 8)       { hh = 0; vv = kl >> 1; }
    else if (kl < 16) { hh = 1; vv = (kl - 8) >> 1; }
    else if (kl < 24) { hh = 0; vv = 4 + ((kl - 16) >> 1); }
    else              { hh = 1; vv = 4 + ((kl - 24) >> 1); }
    slot = (vv << 1) | (kl & 1);
}

// inclusive wave32 scan
__device__ __forceinline__ float wave_iscan(float v, int lane) {
    #pragma unroll
    for (int off = 1; off < 32; off <<= 1) {
        float n = __shfl_up(v, off, 32);
        if (lane >= off) v += n;
    }
    return v;
}

// ---------------------------------------------------------------------------
// Kernel 0: one-time w1 (f32, 1536x128) -> bf16 in B-fragment-tiled order
// wfrag element addr = ((kstep*8 + ntile)*32 + lane)*16 + slot   (384 KB total)
// ---------------------------------------------------------------------------
__global__ void k_w1_pack(const float* __restrict__ w1,
                          unsigned short* __restrict__ wfrag) {
    int idx = blockIdx.x * 256 + threadIdx.x;   // 0 .. 1536*128-1
    int k = idx >> 7;
    int n = idx & 127;
    int kstep = k >> 5, kl = k & 31;
    int nt = n >> 4,    nl = n & 15;
    int hh, slot;
    fragcoord(kl, hh, slot);
    wfrag[(((size_t)kstep * 8 + nt) * 32 + (hh << 4) + nl) * 16 + slot] =
        f2bf(w1[(size_t)k * Aq + n]);
}

// ---------------------------------------------------------------------------
// Kernel 1: causal mean/std per (b,c) row; pack [x|mean|std] bf16 directly
// into A-fragment-tiled order:
// afrag addr = ((mtile*NKS + kstep)*32 + lane)*16 + slot
// one wave per (b,c) row; chunked wave scan with carry
// ---------------------------------------------------------------------------
__global__ void k_stats_pack(const float* __restrict__ x,
                             const int* __restrict__ lengths,
                             unsigned short* __restrict__ afrag) {
    int lane = threadIdx.x & 31;
    int wid  = threadIdx.x >> 5;
    int row  = blockIdx.x * 8 + wid;       // 0 .. B*C-1
    int b = row / Cq, c = row % Cq;
    int len = lengths[b];
    const float* xr = x + ((size_t)b * Cq + c) * Tq;

    // frag coords for this channel's 3 K positions (c, 512+c, 1024+c): loop-invariant
    int kstep_q[3], laneoff_q[3], slot_q[3];
    #pragma unroll
    for (int q = 0; q < 3; q++) {
        int k = q * Cq + c;
        int hh, slot;
        fragcoord(k & 31, hh, slot);
        kstep_q[q] = k >> 5;
        laneoff_q[q] = hh << 4;
        slot_q[q] = slot;
    }

    float carry1 = 0.f, carry2 = 0.f;
    for (int t0 = 0; t0 < Tq; t0 += 32) {
        int   t  = t0 + lane;
        float xv = xr[t];
        float m  = (t < len) ? 1.f : 0.f;
        float xm = xv * m;
        float s1 = wave_iscan(xm, lane);
        float s2 = wave_iscan(xm * xv, lane);
        float S1 = carry1 + s1;
        float S2 = carry2 + s2;
        carry1 += __shfl(s1, 31, 32);
        carry2 += __shfl(s2, 31, 32);

        float cnt  = (float)(t + 1 < len ? t + 1 : len);
        cnt        = fmaxf(cnt, 1.f);
        float mean = S1 / cnt;
        float var  = S2 / cnt - mean * mean;
        float stdv = sqrtf(fmaxf(var, EPSq));

        int grow  = b * Tq + t;            // global GEMM row
        int mtile = grow >> 4, nl = grow & 15;
        size_t tbase = (size_t)mtile * NKS * 512;   // 512 = 32 lanes * 16
        float vals[3] = { xv, mean, stdv };
        #pragma unroll
        for (int q = 0; q < 3; q++)
            afrag[tbase + ((size_t)kstep_q[q] * 32 + laneoff_q[q] + nl) * 16 + slot_q[q]]
                = f2bf(vals[q]);
    }
}

// ---------------------------------------------------------------------------
// Kernel 2: scores = tanh(attn_in @ w1 + b1) @ w2 + b2 via bf16 WMMA
// 8 waves/block, each wave = one 16-row M tile, full N=128 in 8 accumulators.
// All operands pre-swizzled: pure coalesced global b128 streams, no LDS,
// no barriers. w1 frags (384 KB) stay L2-resident.
// ---------------------------------------------------------------------------
__global__ void k_attn_gemm(const unsigned short* __restrict__ afrag_g,
                            const unsigned short* __restrict__ wfrag,
                            const float* __restrict__ b1,
                            const float* __restrict__ w2,
                            const float* __restrict__ b2,
                            float* __restrict__ scores) {
    int lane = threadIdx.x & 31, wid = threadIdx.x >> 5;
    int half = lane >> 4, nl = lane & 15;
    int mtile = blockIdx.x * 8 + wid;

    // v8bf units: one frag-row (32 lanes * 16 bf16) = 64 v8bf
    const v8bf* ap = (const v8bf*)(afrag_g) + ((size_t)mtile * NKS * 32 + lane) * 2;
    const v8bf* bp = (const v8bf*)(wfrag) + (size_t)lane * 2;

    v8f acc[8];
    #pragma unroll
    for (int i = 0; i < 8; i++) acc[i] = (v8f){};

    for (int ks = 0; ks < NKS; ks++) {
        __builtin_prefetch((const void*)(ap + 128), 0, 1);   // 2 k-steps ahead
        v8bf alo = ap[0];
        v8bf ahi = ap[1];
        v16bf af;
        #pragma unroll
        for (int j = 0; j < 8; j++) { af[j] = alo[j]; af[8 + j] = ahi[j]; }

        const v8bf* bks = bp + (size_t)ks * 512;             // ks stride = 8*64
        #pragma unroll
        for (int nt = 0; nt < 8; nt++) {
            v8bf blo = bks[nt * 64];
            v8bf bhi = bks[nt * 64 + 1];
            v16bf bf;
            #pragma unroll
            for (int j = 0; j < 8; j++) { bf[j] = blo[j]; bf[8 + j] = bhi[j]; }
            acc[nt] = __builtin_amdgcn_wmma_f32_16x16x32_bf16(
                false, af, false, bf, (short)0, acc[nt], false, false);
        }
        ap += 64;
    }

    // fused epilogue: +b1 -> tanh -> *w2 -> row reduce -> +b2
    float sp[8];
    #pragma unroll
    for (int v = 0; v < 8; v++) sp[v] = 0.f;
    #pragma unroll
    for (int nt = 0; nt < 8; nt++) {
        int n = nt * 16 + nl;
        float b1n = b1[n], w2n = w2[n];
        #pragma unroll
        for (int v = 0; v < 8; v++)
            sp[v] += tanhf(acc[nt][v] + b1n) * w2n;
    }
    #pragma unroll
    for (int v = 0; v < 8; v++) {
        #pragma unroll
        for (int off = 1; off < 16; off <<= 1)
            sp[v] += __shfl_xor(sp[v], off, 32);
    }
    if (nl == 0) {
        float b2v = b2[0];
        #pragma unroll
        for (int v = 0; v < 8; v++)
            scores[mtile * 16 + half * 8 + v] = sp[v] + b2v;   // M = half*8+v
    }
}

// ---------------------------------------------------------------------------
// Kernel 3: per batch: masked max, e = exp(s-max)*mask, cum_e = cumsum(e)
// one wave per batch
// ---------------------------------------------------------------------------
__global__ void k_softmax_prep(const float* __restrict__ scores,
                               const int* __restrict__ lengths,
                               float* __restrict__ e,
                               float* __restrict__ cume) {
    int b = blockIdx.x;
    int lane = threadIdx.x;
    int len = lengths[b];
    const float* s = scores + (size_t)b * Tq;

    float mx = -3.4e38f;
    for (int t0 = 0; t0 < Tq; t0 += 32) {
        int t = t0 + lane;
        float v = (t < len) ? s[t] : -3.4e38f;
        mx = fmaxf(mx, v);
    }
    #pragma unroll
    for (int off = 16; off >= 1; off >>= 1)
        mx = fmaxf(mx, __shfl_xor(mx, off, 32));

    float carry = 0.f;
    for (int t0 = 0; t0 < Tq; t0 += 32) {
        int t = t0 + lane;
        float ev = (t < len) ? expf(s[t] - mx) : 0.f;
        float sc = wave_iscan(ev, lane);
        e[(size_t)b * Tq + t]    = ev;
        cume[(size_t)b * Tq + t] = carry + sc;
        carry += __shfl(sc, 31, 32);
    }
}

// ---------------------------------------------------------------------------
// Kernel 4: per (b,c): scans of e*x, e*x^2 -> w_mean/w_std, masked average
// one wave per (b,c) row
// ---------------------------------------------------------------------------
__global__ void k_final(const float* __restrict__ x,
                        const int* __restrict__ lengths,
                        const float* __restrict__ e,
                        const float* __restrict__ cume,
                        float* __restrict__ out) {
    int lane = threadIdx.x & 31;
    int wid  = threadIdx.x >> 5;
    int row  = blockIdx.x * 8 + wid;
    int b = row / Cq, c = row % Cq;
    int len = lengths[b];
    const float* xr = x + ((size_t)b * Cq + c) * Tq;
    const float* er = e + (size_t)b * Tq;
    const float* cr = cume + (size_t)b * Tq;

    float carry1 = 0.f, carry2 = 0.f;
    float am = 0.f, as = 0.f;
    for (int t0 = 0; t0 < Tq; t0 += 32) {
        int   t  = t0 + lane;
        float xv = xr[t];
        float ev = er[t];
        float s1 = wave_iscan(ev * xv, lane);
        float s2 = wave_iscan(ev * xv * xv, lane);
        float S1 = carry1 + s1;
        float S2 = carry2 + s2;
        carry1 += __shfl(s1, 31, 32);
        carry2 += __shfl(s2, 31, 32);
        if (t < len) {
            float inv = 1.f / cr[t];
            float wm  = S1 * inv;
            float wv  = S2 * inv - wm * wm;
            float wsd = sqrtf(fmaxf(wv, EPSq));
            am += wm;
            as += wsd;
        }
    }
    #pragma unroll
    for (int off = 16; off >= 1; off >>= 1) {
        am += __shfl_xor(am, off, 32);
        as += __shfl_xor(as, off, 32);
    }
    if (lane == 0) {
        float norm = fmaxf((float)len, EPSq);
        out[(size_t)b * (2 * Cq) + c]      = am / norm;
        out[(size_t)b * (2 * Cq) + Cq + c] = as / norm;
    }
}

// ---------------------------------------------------------------------------
extern "C" void kernel_launch(void* const* d_in, const int* in_sizes, int n_in,
                              void* d_out, int out_size, void* d_ws, size_t ws_size,
                              hipStream_t stream) {
    const float* x       = (const float*)d_in[0];
    const int*   lengths = (const int*)d_in[1];
    const float* w1      = (const float*)d_in[2];
    const float* b1      = (const float*)d_in[3];
    const float* w2      = (const float*)d_in[4];
    const float* b2      = (const float*)d_in[5];
    float* out = (float*)d_out;

    char* ws = (char*)d_ws;
    unsigned short* afrag = (unsigned short*)ws;              // 48 MB bf16 (A, tiled)
    size_t off = (size_t)Bq * Tq * K3C * sizeof(unsigned short);
    unsigned short* wfrag = (unsigned short*)(ws + off);      // 384 KB bf16 (B, tiled)
    off += (size_t)K3C * Aq * sizeof(unsigned short);
    float* scores = (float*)(ws + off); off += (size_t)Bq * Tq * sizeof(float);
    float* e      = (float*)(ws + off); off += (size_t)Bq * Tq * sizeof(float);
    float* cume   = (float*)(ws + off);

    k_w1_pack     <<<(K3C * Aq) / 256, 256, 0, stream>>>(w1, wfrag);
    k_stats_pack  <<<(Bq * Cq) / 8, 256, 0, stream>>>(x, lengths, afrag);
    k_attn_gemm   <<<(Bq * Tq / 16) / 8, 256, 0, stream>>>(afrag, wfrag, b1, w2, b2, scores);
    k_softmax_prep<<<Bq, 32, 0, stream>>>(scores, lengths, e, cume);
    k_final       <<<(Bq * Cq) / 8, 256, 0, stream>>>(x, lengths, e, cume, out);
}
